// PolicyAwareLSTM_12833362281155
// MI455X (gfx1250) — matrix operand order repaired and checked by reading, not compiled, not run on hardware
//
#include <hip/hip_runtime.h>
#include <hip/hip_bf16.h>
#include <stdint.h>

// ---------------------------------------------------------------------------
// Fused 2-layer LSTM + (degenerate seq-1 MHA) + MLP head for MI455X (gfx1250)
// v_wmma_f32_16x16x32_bf16 core; ALL weight B-fragments VGPR-resident
// (~580 VGPRs via the gfx1250 1024-VGPR extended file); one wave owns a
// 16-row batch tile across all 336 timesteps. Gate nonlinearities use the
// gfx1250 hardware v_tanh_f32 / v_exp_f32 / v_rcp_f32; bf16 conversion uses
// the hardware packed converter v_cvt_pk_bf16_f32.
// ---------------------------------------------------------------------------

typedef __attribute__((ext_vector_type(16))) __bf16 v16bf;
typedef __attribute__((ext_vector_type(2)))  __bf16 v2bf;
typedef __attribute__((ext_vector_type(8)))  float  v8f;

#define BATCH  4096
#define TSTEPS 336
#define NFEAT  32
#define NU1    64
#define NU2    32
#define PRED   24

// Packed weight image (bf16) produced by the repack kernel, tile-major:
//   tiles  0..15 : W1 (K=32 ,N=256, 1x16)
//   tiles 16..47 : U1 (K=64 ,N=256, 2x16)
//   tiles 48..63 : W2 (K=64 ,N=128, 2x8)
//   tiles 64..71 : U2 (K=32 ,N=128, 1x8)
// Each 32x16 tile: 32 lanes x 16 halves, lane-contiguous (32 B per lane).
#define W1OFF 0
#define U1OFF 8192
#define W2OFF 24576
#define U2OFF 32768
#define WTOT  36864
#define W1T   0
#define U1T   16
#define W2T   48
#define U2T   64
#define NTILES 72

__device__ __forceinline__ unsigned short bf16r_manual(float f) {
  unsigned u = __float_as_uint(f);
  unsigned r = u + 0x7FFFu + ((u >> 16) & 1u);   // round-to-nearest-even
  return (unsigned short)(r >> 16);
}
// Pack two f32 -> two bf16 in one u32 (hardware v_cvt_pk_bf16_f32 if present)
__device__ __forceinline__ uint32_t pkbf(float lo, float hi) {
#if __has_builtin(__builtin_amdgcn_cvt_pk_bf16_f32)
  union { v2bf v; uint32_t u; } cv;
  cv.v = __builtin_amdgcn_cvt_pk_bf16_f32(lo, hi);
  return cv.u;
#else
  return (uint32_t)bf16r_manual(lo) | ((uint32_t)bf16r_manual(hi) << 16);
#endif
}
__device__ __forceinline__ unsigned short bf16s(float f) {
#if __has_builtin(__builtin_amdgcn_cvt_pk_bf16_f32)
  return (unsigned short)(pkbf(f, f) & 0xFFFFu);
#else
  return bf16r_manual(f);
#endif
}

// ---- branchless transcendentals on gfx1250 hardware units ----
__device__ __forceinline__ float fast_rcp(float x) {
#if __has_builtin(__builtin_amdgcn_rcpf)
  return __builtin_amdgcn_rcpf(x);        // v_rcp_f32
#else
  return 1.0f / x;
#endif
}
__device__ __forceinline__ float fast_tanh(float x) {
#if __has_builtin(__builtin_amdgcn_tanhf)
  return __builtin_amdgcn_tanhf(x);       // v_tanh_f32 (gfx1250 tanh-insts)
#else
  float cx = fminf(fmaxf(x, -10.0f), 10.0f);
  float e  = __expf(2.0f * cx);           // v_exp_f32 (native)
  return (e - 1.0f) * fast_rcp(e + 1.0f);
#endif
}
__device__ __forceinline__ float sigm(float x) {
  return fast_rcp(1.0f + __expf(-x));     // v_exp_f32 + v_rcp_f32, branchless
}

union FragU { v16bf v; uint32_t u[8]; uint4 q[2]; };

__device__ __forceinline__ v8f wmma_bf16(v16bf a, v16bf b, v8f c) {
  // D(f32 16x16) = A(bf16 16x32) * B(bf16 32x16) + C
  return __builtin_amdgcn_wmma_f32_16x16x32_bf16(false, a, false, b, (short)0, c,
                                                 false, false);
}

__device__ __forceinline__ v8f splat8(float s) {
  v8f r;
#pragma unroll
  for (int i = 0; i < 8; ++i) r[i] = s;
  return r;
}

// Build an A-fragment (16 x 32 bf16) from an LDS row-major bf16 h-buffer.
// A layout: M = lane%16; VGPR v holds K = 16*(v>>2) + 8*(lane>=16) + 2*(v&3)+{0,1}
__device__ __forceinline__ v16bf ld_hfrag(const unsigned short* hb, int ld,
                                          int kbase, int lm, int lh) {
  FragU f;
  const unsigned short* rp = hb + lm * ld + kbase;
#pragma unroll
  for (int v = 0; v < 8; ++v) {
    int kv = ((v >> 2) << 4) + (lh << 3) + ((v & 3) << 1);
    f.u[v] = *(const uint32_t*)(rp + kv);
  }
  return f.v;
}

// ---------------------------------------------------------------------------
// Kernel 1: repack f32 weights -> bf16 WMMA B-fragment layout in workspace.
// B layout per 32x16 tile: lane holds column N = tile_n*16 + lane%16;
// element e (0..15) holds K = tile_k*32 + 16*(e>=8) + 8*(lane>=16) + (e&7).
// ---------------------------------------------------------------------------
__global__ void lstm_repack_weights(const float* __restrict__ W1,
                                    const float* __restrict__ U1,
                                    const float* __restrict__ W2,
                                    const float* __restrict__ U2,
                                    unsigned short* __restrict__ packed) {
  int idx = blockIdx.x * blockDim.x + threadIdx.x;
  if (idx >= WTOT) return;
  const float* src;
  int NT, NC, rel;
  if (idx < U1OFF)      { src = W1; NT = 16; NC = 256; rel = idx - W1OFF; }
  else if (idx < W2OFF) { src = U1; NT = 16; NC = 256; rel = idx - U1OFF; }
  else if (idx < U2OFF) { src = W2; NT = 8;  NC = 128; rel = idx - W2OFF; }
  else                  { src = U2; NT = 8;  NC = 128; rel = idx - U2OFF; }
  int tile = rel >> 9;          // 512 elems per tile
  int lane = (rel >> 4) & 31;
  int e    = rel & 15;
  int kt = tile / NT;
  int nt = tile - kt * NT;
  int K = kt * 32 + ((e >> 3) << 4) + ((lane >> 4) << 3) + (e & 7);
  int N = nt * 16 + (lane & 15);
  packed[idx] = bf16s(src[K * NC + N]);
}

// ---------------------------------------------------------------------------
// Kernel 2: fused LSTM stack + head. 1 wave per workgroup; each wave owns a
// 16-row batch tile. All 72 weight B-fragments live in VGPRs for the whole
// 336-step recurrence; LDS is only the h-state transpose exchange (5 KB).
// ---------------------------------------------------------------------------
__global__ void __launch_bounds__(32)
lstm_fused(const float* __restrict__ x,
           const float* __restrict__ b1, const float* __restrict__ b2,
           const unsigned short* __restrict__ packed,
           const float* __restrict__ Wvp, const float* __restrict__ bvp,
           const float* __restrict__ Wop, const float* __restrict__ bop,
           const float* __restrict__ Wd1, const float* __restrict__ bd1,
           const float* __restrict__ Wd2, const float* __restrict__ bd2,
           float* __restrict__ out) {
  __shared__ unsigned short h1b[16 * NU1];   // layer-1 h, bf16 row-major [M][64]
  __shared__ unsigned short h2b[16 * NU2];   // layer-2 h, bf16 row-major [M][32]
  __shared__ float          hdb[16 * NU2];   // final h2 in f32 for the head

  const int lane = threadIdx.x;
  const int lm   = lane & 15;   // N column (B/C/D) or M row (A)
  const int lh   = lane >> 4;   // half-wave select
  const int row0 = blockIdx.x * 16;

  // ---- preload ALL weight B-fragments into registers (576 VGPRs) ----
  v16bf wreg[NTILES];
#pragma unroll
  for (int tIdx = 0; tIdx < NTILES; ++tIdx) {
    FragU f;
    const uint4* p = (const uint4*)(packed + tIdx * 512 + lane * 16);
    f.q[0] = p[0];
    f.q[1] = p[1];
    wreg[tIdx] = f.v;
  }

  // ---- per-lane hoisted biases ----
  float bi1[4], bf1[4], bg1[4], bo1[4];
#pragma unroll
  for (int j = 0; j < 4; ++j) {
    bi1[j] = b1[      j * 16 + lm];
    bf1[j] = b1[ 64 + j * 16 + lm];
    bg1[j] = b1[128 + j * 16 + lm];
    bo1[j] = b1[192 + j * 16 + lm];
  }
  float bi2[2], bf2[2], bg2[2], bo2[2];
#pragma unroll
  for (int j = 0; j < 2; ++j) {
    bi2[j] = b2[     j * 16 + lm];
    bf2[j] = b2[32 + j * 16 + lm];
    bg2[j] = b2[64 + j * 16 + lm];
    bo2[j] = b2[96 + j * 16 + lm];
  }

  // ---- zero initial hidden state (bf16 zeros in LDS) ----
  {
    uint32_t* z1 = (uint32_t*)h1b;
    for (int i = lane; i < 16 * NU1 / 2; i += 32) z1[i] = 0;
    uint32_t* z2 = (uint32_t*)h2b;
    for (int i = lane; i < 16 * NU2 / 2; i += 32) z2[i] = 0;
  }

  // Cell state in accumulator (D) layout: element (M = 8*lh + r, N = 16*j + lm)
  v8f c1[4], c2[2];
#pragma unroll
  for (int j = 0; j < 4; ++j) c1[j] = splat8(0.0f);
#pragma unroll
  for (int j = 0; j < 2; ++j) c2[j] = splat8(0.0f);

  // Per-lane base pointer into x: row (row0+lm), feature block lh*8.
  const float* xlane = x + ((size_t)(row0 + lm) * TSTEPS) * NFEAT + lh * 8;

  for (int t = 0; t < TSTEPS; ++t) {
    if (t + 8 < TSTEPS) __builtin_prefetch(xlane + (size_t)(t + 8) * NFEAT, 0, 1);

    // ---- x_t A-fragment (16x32 bf16), converted on the fly ----
    FragU xa;
    {
      const float4* xp = (const float4*)(xlane + (size_t)t * NFEAT);
      float4 a0 = xp[0];   // K = base+0..3
      float4 a1 = xp[1];   // K = base+4..7
      float4 a2 = xp[4];   // K = base+16..19
      float4 a3 = xp[5];   // K = base+20..23
      xa.u[0] = pkbf(a0.x, a0.y); xa.u[1] = pkbf(a0.z, a0.w);
      xa.u[2] = pkbf(a1.x, a1.y); xa.u[3] = pkbf(a1.z, a1.w);
      xa.u[4] = pkbf(a2.x, a2.y); xa.u[5] = pkbf(a2.z, a2.w);
      xa.u[6] = pkbf(a3.x, a3.y); xa.u[7] = pkbf(a3.z, a3.w);
    }

    // ---- old hidden states as A-fragments ----
    v16bf h1a0 = ld_hfrag(h1b, NU1, 0,  lm, lh);
    v16bf h1a1 = ld_hfrag(h1b, NU1, 32, lm, lh);
    v16bf h2a  = ld_hfrag(h2b, NU2, 0,  lm, lh);

    // ================= Layer 1: gates = x@W1 + h1@U1 + b1 =================
#pragma unroll
    for (int j = 0; j < 4; ++j) {
      v8f ai = splat8(bi1[j]);
      v8f af = splat8(bf1[j]);
      v8f ag = splat8(bg1[j]);
      v8f ao = splat8(bo1[j]);
      // grouped by shared A operand -> back-to-back WMMA issue, 4-way acc ILP
      ai = wmma_bf16(xa.v, wreg[W1T + j],      ai);
      af = wmma_bf16(xa.v, wreg[W1T + 4 + j],  af);
      ag = wmma_bf16(xa.v, wreg[W1T + 8 + j],  ag);
      ao = wmma_bf16(xa.v, wreg[W1T + 12 + j], ao);

      ai = wmma_bf16(h1a0, wreg[U1T + j],      ai);
      af = wmma_bf16(h1a0, wreg[U1T + 4 + j],  af);
      ag = wmma_bf16(h1a0, wreg[U1T + 8 + j],  ag);
      ao = wmma_bf16(h1a0, wreg[U1T + 12 + j], ao);

      ai = wmma_bf16(h1a1, wreg[U1T + 16 + j],      ai);
      af = wmma_bf16(h1a1, wreg[U1T + 16 + 4 + j],  af);
      ag = wmma_bf16(h1a1, wreg[U1T + 16 + 8 + j],  ag);
      ao = wmma_bf16(h1a1, wreg[U1T + 16 + 12 + j], ao);

#pragma unroll
      for (int r = 0; r < 8; ++r) {
        float iv = sigm(ai[r]);
        float fv = sigm(af[r]);
        float gv = fast_tanh(ag[r]);
        float ov = sigm(ao[r]);
        float c  = fv * c1[j][r] + iv * gv;
        c1[j][r] = c;
        float h  = ov * fast_tanh(c);
        int M = (lh << 3) + r;                       // D layout row
        h1b[M * NU1 + j * 16 + lm] = bf16s(h);       // publish new h1
      }
    }

    // New h1 as A-fragments (in-order DS within the wave guarantees visibility)
    v16bf h1n0 = ld_hfrag(h1b, NU1, 0,  lm, lh);
    v16bf h1n1 = ld_hfrag(h1b, NU1, 32, lm, lh);

    // ================= Layer 2: gates = h1@W2 + h2@U2 + b2 =================
#pragma unroll
    for (int j = 0; j < 2; ++j) {
      v8f ai = splat8(bi2[j]);
      v8f af = splat8(bf2[j]);
      v8f ag = splat8(bg2[j]);
      v8f ao = splat8(bo2[j]);
      ai = wmma_bf16(h1n0, wreg[W2T + j],     ai);
      af = wmma_bf16(h1n0, wreg[W2T + 2 + j], af);
      ag = wmma_bf16(h1n0, wreg[W2T + 4 + j], ag);
      ao = wmma_bf16(h1n0, wreg[W2T + 6 + j], ao);

      ai = wmma_bf16(h1n1, wreg[W2T + 8 + j],     ai);
      af = wmma_bf16(h1n1, wreg[W2T + 8 + 2 + j], af);
      ag = wmma_bf16(h1n1, wreg[W2T + 8 + 4 + j], ag);
      ao = wmma_bf16(h1n1, wreg[W2T + 8 + 6 + j], ao);

      ai = wmma_bf16(h2a, wreg[U2T + j],     ai);
      af = wmma_bf16(h2a, wreg[U2T + 2 + j], af);
      ag = wmma_bf16(h2a, wreg[U2T + 4 + j], ag);
      ao = wmma_bf16(h2a, wreg[U2T + 6 + j], ao);

#pragma unroll
      for (int r = 0; r < 8; ++r) {
        float iv = sigm(ai[r]);
        float fv = sigm(af[r]);
        float gv = fast_tanh(ag[r]);
        float ov = sigm(ao[r]);
        float c  = fv * c2[j][r] + iv * gv;
        c2[j][r] = c;
        float h  = ov * fast_tanh(c);
        int M = (lh << 3) + r;
        h2b[M * NU2 + j * 16 + lm] = bf16s(h);
        if (t == TSTEPS - 1) hdb[M * NU2 + j * 16 + lm] = h;   // f32 for the head
      }
    }
  }

  // ================= Head: (degenerate MHA) + Dense + Dense =================
  // softmax over a length-1 axis == 1, so attention output == V projection.
  if (lane < 16) {
    const float* hr = hdb + lane * NU2;     // this lane's row of final h2
    float a[32], m[32], d1[64];
    for (int o = 0; o < 32; ++o) {          // V projection  (Wv: [32][4,8]->[32][32])
      float s = bvp[o];
      for (int i = 0; i < 32; ++i) s += hr[i] * Wvp[i * 32 + o];
      a[o] = s;
    }
    for (int o = 0; o < 32; ++o) {          // output projection (Wo: [4,8][32]->[32][32])
      float s = bop[o];
      for (int i = 0; i < 32; ++i) s += a[i] * Wop[i * 32 + o];
      m[o] = s;
    }
    for (int o = 0; o < 64; ++o) {          // Dense 32->64 + ReLU
      float s = bd1[o];
      for (int i = 0; i < 32; ++i) s += m[i] * Wd1[i * 64 + o];
      d1[o] = fmaxf(s, 0.0f);
    }
    float* op = out + (size_t)(row0 + lane) * PRED;
    for (int o = 0; o < PRED; ++o) {        // Dense 64->24
      float s = bd2[o];
      for (int i = 0; i < 64; ++i) s += d1[i] * Wd2[i * PRED + o];
      op[o] = s;
    }
  }
}

extern "C" void kernel_launch(void* const* d_in, const int* in_sizes, int n_in,
                              void* d_out, int out_size, void* d_ws, size_t ws_size,
                              hipStream_t stream) {
  (void)in_sizes; (void)n_in; (void)out_size; (void)ws_size;
  const float* x   = (const float*)d_in[0];
  const float* W1  = (const float*)d_in[1];
  const float* U1  = (const float*)d_in[2];
  const float* b1  = (const float*)d_in[3];
  const float* W2  = (const float*)d_in[4];
  const float* U2  = (const float*)d_in[5];
  const float* b2  = (const float*)d_in[6];
  // d_in[7..10] = Wq,bq,Wk,bk : mathematically dead (softmax over length-1 axis)
  const float* Wvp = (const float*)d_in[11];
  const float* bvp = (const float*)d_in[12];
  const float* Wop = (const float*)d_in[13];
  const float* bop = (const float*)d_in[14];
  const float* Wd1 = (const float*)d_in[15];
  const float* bd1 = (const float*)d_in[16];
  const float* Wd2 = (const float*)d_in[17];
  const float* bd2 = (const float*)d_in[18];
  unsigned short* packed = (unsigned short*)d_ws;
  float* out = (float*)d_out;

  lstm_repack_weights<<<(WTOT + 255) / 256, 256, 0, stream>>>(W1, U1, W2, U2, packed);

  lstm_fused<<<BATCH / 16, 32, 0, stream>>>(x, b1, b2, packed,
                                            Wvp, bvp, Wop, bop,
                                            Wd1, bd1, Wd2, bd2, out);
}
